// STU_90787018702948
// MI455X (gfx1250) — matrix operand order repaired and checked by reading, not compiled, not run on hardware
//
#include <hip/hip_runtime.h>

// Problem constants (from reference)
#define BB 32
#define NN 512
#define FF 512
#define HH 8
#define DD 64
#define KTOT 512   // inner K of both big GEMMs
#define NOUT1 2048 // uvqk output width

typedef unsigned short u16;
typedef __attribute__((ext_vector_type(8)))  unsigned short v8us;
typedef __attribute__((ext_vector_type(16))) unsigned short v16us;
typedef __attribute__((ext_vector_type(16))) __bf16        v16bf;
typedef __attribute__((ext_vector_type(8)))  float         v8f;

static __device__ __forceinline__ u16 f2bf(float f) {
  unsigned u = __builtin_bit_cast(unsigned, f);
  u += 0x7FFFu + ((u >> 16) & 1u);          // round-to-nearest-even
  return (u16)(u >> 16);
}
static __device__ __forceinline__ float siluf(float x) {
  return x * (1.0f / (1.0f + __expf(-x)));
}
// A fragment (16x32 bf16): lane = row; elems 0..7 = K[half*8 .. +7], 8..15 = K[16+half*8 .. +7]
static __device__ __forceinline__ v16bf load_a_frag(const u16* p, int half) {
  v16us t;
  *(v8us*)&t         = *(const v8us*)(p + half * 8);
  *(((v8us*)&t) + 1) = *(const v8us*)(p + 16 + half * 8);
  return __builtin_bit_cast(v16bf, t);
}
// B fragment (32x16 bf16): lane = column; elems 0..7 = K[half*8 .. +7], 8..15 = K[16+half*8 .. +7]
static __device__ __forceinline__ v16bf load_b_frag(const u16* p, int half) {
  v16us t;
  *(v8us*)&t         = *(const v8us*)(p + half * 8);
  *(((v8us*)&t) + 1) = *(const v8us*)(p + 16 + half * 8);
  return __builtin_bit_cast(v16bf, t);
}
#define WMMA_BF16(a, b, c) \
  __builtin_amdgcn_wmma_f32_16x16x32_bf16(false, (a), false, (b), (short)0, (c), false, false)

// ---------------- LayerNorm(x) * tl -> bf16 ----------------
__global__ void __launch_bounds__(256) k_ln_x(const float* __restrict__ x,
                                              const float* __restrict__ tl,
                                              u16* __restrict__ xn) {
  int row = blockIdx.x, t = threadIdx.x;
  float2 v = ((const float2*)(x + (size_t)row * FF))[t];
  float s = v.x + v.y, ss = v.x * v.x + v.y * v.y;
  for (int o = 16; o; o >>= 1) { s += __shfl_xor(s, o); ss += __shfl_xor(ss, o); }
  __shared__ float rs[8], rss[8];
  int wv = t >> 5;
  if ((t & 31) == 0) { rs[wv] = s; rss[wv] = ss; }
  __syncthreads();
  s = 0.f; ss = 0.f;
#pragma unroll
  for (int i = 0; i < 8; i++) { s += rs[i]; ss += rss[i]; }
  float mean = s * (1.0f / FF);
  float var  = ss * (1.0f / FF) - mean * mean;
  float rstd = __frsqrt_rn(var + 1e-6f);
  float tm = tl[row];
  unsigned lo = f2bf((v.x - mean) * rstd * tm);
  unsigned hi = f2bf((v.y - mean) * rstd * tm);
  ((unsigned*)xn)[(size_t)row * (FF / 2) + t] = lo | (hi << 16);
}

// ---------------- transpose weight f32(K x Nc) -> bf16(Nc x K) ----------------
__global__ void k_tr_w(const float* __restrict__ w, u16* __restrict__ wt,
                       int krows, int ncols) {
  int idx = blockIdx.x * blockDim.x + threadIdx.x;
  if (idx >= krows * ncols) return;
  int n = idx / krows, k = idx - n * krows;
  wt[idx] = f2bf(w[(size_t)k * ncols + n]);
}

// ---------------- g = silu(Xn @ W) * tl ; split into u,v,q,k ----------------
__global__ void __launch_bounds__(256) k_gemm_uvqk(
    const u16* __restrict__ xn, const u16* __restrict__ wt,
    const float* __restrict__ tl, float* __restrict__ uu,
    u16* __restrict__ vt, u16* __restrict__ qb, u16* __restrict__ kb) {
  int wv = threadIdx.x >> 5, lane = threadIdx.x & 31;
  int half = lane >> 4, l15 = lane & 15;
  int row0 = blockIdx.x * 128 + wv * 16;
  int col0 = blockIdx.y * 64;
  v8f acc[4];
#pragma unroll
  for (int i = 0; i < 4; i++) acc[i] = (v8f)(0.0f);
  const u16* arow = xn + (size_t)(row0 + l15) * KTOT;
#pragma unroll 4
  for (int kk = 0; kk < KTOT; kk += 32) {
    __builtin_prefetch(arow + kk + 128, 0, 1);
    v16bf a = load_a_frag(arow + kk, half);
#pragma unroll
    for (int nt = 0; nt < 4; nt++) {
      v16bf b = load_b_frag(wt + (size_t)(col0 + nt * 16 + l15) * KTOT + kk, half);
      acc[nt] = WMMA_BF16(a, b, acc[nt]);
    }
  }
  float tlr[8];
#pragma unroll
  for (int i = 0; i < 8; i++) tlr[i] = tl[row0 + half * 8 + i];
  int sec = col0 >> 9;          // 0:u 1:v 2:q 3:k (each section 512 wide)
  int cb = col0 & 511;
#pragma unroll
  for (int nt = 0; nt < 4; nt++) {
    int c = cb + nt * 16 + l15;
#pragma unroll
    for (int i = 0; i < 8; i++) {
      int row = row0 + half * 8 + i;
      float s = siluf(acc[nt][i]) * tlr[i];
      if (sec == 0) {
        uu[(size_t)row * FF + c] = s;
      } else if (sec == 1) {
        int bI = row >> 9, n = row & 511;
        int h = c >> 6, dl = c & 63;
        // V stored transposed per (b,h): vt[((b*H+h)*D + dl)*N + n]
        vt[(size_t)((bI * HH + h) * DD + dl) * NN + n] = f2bf(s);
      } else if (sec == 2) {
        qb[(size_t)row * FF + c] = f2bf(s);
      } else {
        kb[(size_t)row * FF + c] = f2bf(s);
      }
    }
  }
}

// ---------------- fused attention: O = (silu(QK^T + rel) * mask / N) @ V ----------------
__global__ void __launch_bounds__(256) k_attn(
    const u16* __restrict__ qb, const u16* __restrict__ kb,
    const u16* __restrict__ vt, const float* __restrict__ tl,
    const float* __restrict__ tw, const float* __restrict__ pw,
    const int* __restrict__ ts, float* __restrict__ ao) {
  __shared__ __align__(32) u16 smem[8 * 16 * 32];  // per-wave private 16x32 bf16 S tile
  int wv = threadIdx.x >> 5, lane = threadIdx.x & 31;
  int half = lane >> 4, l15 = lane & 15;
  int bI = blockIdx.z, h = blockIdx.y;
  int qr0 = blockIdx.x * 128 + wv * 16;
  u16* sw = smem + wv * 16 * 32;

  // q A-fragments for DA=64 (two K=32 slabs), reused across all key blocks
  const u16* qrow_p = qb + ((size_t)(bI * NN + qr0 + l15) * FF + h * DD);
  v16bf aq0 = load_a_frag(qrow_p, half);
  v16bf aq1 = load_a_frag(qrow_p + 32, half);

  float tq[8]; int tsq[8];
#pragma unroll
  for (int i = 0; i < 8; i++) {
    int qr = qr0 + half * 8 + i;
    tq[i]  = tl[bI * NN + qr];
    tsq[i] = ts[bI * (NN + 1) + qr + 1];
  }

  v8f oacc[4];
#pragma unroll
  for (int i = 0; i < 4; i++) oacc[i] = (v8f)(0.0f);

  for (int m0 = 0; m0 < NN; m0 += 32) {
    v8f s0 = (v8f)(0.0f), s1 = (v8f)(0.0f);
#pragma unroll
    for (int kd = 0; kd < DD; kd += 32) {
      v16bf a = (kd == 0) ? aq0 : aq1;
      v16bf bf0 = load_b_frag(kb + ((size_t)(bI * NN + m0 + l15) * FF + h * DD + kd), half);
      v16bf bf1 = load_b_frag(kb + ((size_t)(bI * NN + m0 + 16 + l15) * FF + h * DD + kd), half);
      s0 = WMMA_BF16(a, bf0, s0);
      s1 = WMMA_BF16(a, bf1, s1);
    }
    // bias + silu + causal/timeline mask, re-shape through LDS (same wave: no barrier)
#pragma unroll
    for (int st = 0; st < 2; st++) {
      v8f sv = st ? s1 : s0;
      int key = m0 + st * 16 + l15;
      float tk = tl[bI * NN + key];
      int tsk = ts[bI * (NN + 1) + key];
#pragma unroll
      for (int i = 0; i < 8; i++) {
        int qr = qr0 + half * 8 + i;
        float dt = fabsf((float)(tsq[i] - tsk));
        dt = fmaxf(dt, 1.0f);
        int bk = (int)(__logf(dt) * (1.0f / 0.301f));
        bk = bk < 0 ? 0 : (bk > 128 ? 128 : bk);
        float val = sv[i] + pw[NN - 1 + key - qr] + tw[bk];
        val = siluf(val) * (1.0f / NN);
        val = (key <= qr) ? val * tq[i] * tk : 0.0f;
        sw[(half * 8 + i) * 32 + st * 16 + l15] = f2bf(val);
      }
    }
    // O += S @ V  (V pre-transposed: columns contiguous over keys)
    v16bf as = load_a_frag(sw + l15 * 32, half);
#pragma unroll
    for (int nt = 0; nt < 4; nt++) {
      v16bf bv = load_b_frag(vt + ((size_t)((bI * HH + h) * DD + nt * 16 + l15) * NN + m0), half);
      oacc[nt] = WMMA_BF16(as, bv, oacc[nt]);
    }
  }
#pragma unroll
  for (int nt = 0; nt < 4; nt++)
#pragma unroll
    for (int i = 0; i < 8; i++) {
      int qr = qr0 + half * 8 + i;
      ao[(size_t)(bI * NN + qr) * FF + h * DD + nt * 16 + l15] = oacc[nt][i];
    }
}

// ---------------- o = u * LN(attn_out) * tl -> bf16 ----------------
__global__ void __launch_bounds__(256) k_ln_attn(const float* __restrict__ ao,
                                                 const float* __restrict__ uu,
                                                 const float* __restrict__ tl,
                                                 u16* __restrict__ ob) {
  int row = blockIdx.x, t = threadIdx.x;
  float2 v = ((const float2*)(ao + (size_t)row * FF))[t];
  float s = v.x + v.y, ss = v.x * v.x + v.y * v.y;
  for (int o = 16; o; o >>= 1) { s += __shfl_xor(s, o); ss += __shfl_xor(ss, o); }
  __shared__ float rs[8], rss[8];
  int wv = t >> 5;
  if ((t & 31) == 0) { rs[wv] = s; rss[wv] = ss; }
  __syncthreads();
  s = 0.f; ss = 0.f;
#pragma unroll
  for (int i = 0; i < 8; i++) { s += rs[i]; ss += rss[i]; }
  float mean = s * (1.0f / FF);
  float var  = ss * (1.0f / FF) - mean * mean;
  float rstd = __frsqrt_rn(var + 1e-6f);
  float tm = tl[row];
  float2 uv = ((const float2*)(uu + (size_t)row * FF))[t];
  unsigned lo = f2bf(uv.x * (v.x - mean) * rstd * tm);
  unsigned hi = f2bf(uv.y * (v.y - mean) * rstd * tm);
  ((unsigned*)ob)[(size_t)row * (FF / 2) + t] = lo | (hi << 16);
}

// ---------------- out = o @ out_w + out_b + x ----------------
__global__ void __launch_bounds__(256) k_gemm_out(
    const u16* __restrict__ ob, const u16* __restrict__ wt2,
    const float* __restrict__ x, const float* __restrict__ bias,
    float* __restrict__ out) {
  int wv = threadIdx.x >> 5, lane = threadIdx.x & 31;
  int half = lane >> 4, l15 = lane & 15;
  int row0 = blockIdx.x * 128 + wv * 16;
  int col0 = blockIdx.y * 64;
  v8f acc[4];
#pragma unroll
  for (int i = 0; i < 4; i++) acc[i] = (v8f)(0.0f);
  const u16* arow = ob + (size_t)(row0 + l15) * KTOT;
#pragma unroll 4
  for (int kk = 0; kk < KTOT; kk += 32) {
    __builtin_prefetch(arow + kk + 128, 0, 1);
    v16bf a = load_a_frag(arow + kk, half);
#pragma unroll
    for (int nt = 0; nt < 4; nt++) {
      v16bf b = load_b_frag(wt2 + (size_t)(col0 + nt * 16 + l15) * KTOT + kk, half);
      acc[nt] = WMMA_BF16(a, b, acc[nt]);
    }
  }
#pragma unroll
  for (int nt = 0; nt < 4; nt++) {
    int col = col0 + nt * 16 + l15;
    float bv = bias[col];
#pragma unroll
    for (int i = 0; i < 8; i++) {
      int row = row0 + half * 8 + i;
      out[(size_t)row * FF + col] = acc[nt][i] + bv + x[(size_t)row * FF + col];
    }
  }
}

extern "C" void kernel_launch(void* const* d_in, const int* in_sizes, int n_in,
                              void* d_out, int out_size, void* d_ws, size_t ws_size,
                              hipStream_t stream) {
  const float* x   = (const float*)d_in[0];
  const float* tl  = (const float*)d_in[1];
  // d_in[2] = attn_mask (causal tril; applied analytically as key <= query)
  const float* w1  = (const float*)d_in[3];   // uvqk_proj (512 x 2048)
  const float* w2  = (const float*)d_in[4];   // out_w (512 x 512)
  const float* obv = (const float*)d_in[5];   // out_b
  const float* tw  = (const float*)d_in[6];   // time_weights
  const float* pw  = (const float*)d_in[7];   // pos_weights
  const int*   ts  = (const int*)d_in[8];     // unix_ts (B, N+1)

  char* ws = (char*)d_ws;
  size_t off = 0;
  auto alloc = [&](size_t bytes) -> void* {
    void* p = ws + off;
    off += (bytes + 255) & ~(size_t)255;
    return p;
  };
  const size_t M = (size_t)BB * NN;            // 16384
  u16*   xn  = (u16*)alloc(M * FF * 2);        // normed x, bf16
  u16*   wt1 = (u16*)alloc((size_t)NOUT1 * KTOT * 2);
  u16*   wt2 = (u16*)alloc((size_t)FF * KTOT * 2);
  float* uu  = (float*)alloc(M * FF * 4);      // u gate, f32
  u16*   vt  = (u16*)alloc(M * FF * 2);        // V transposed per (b,h), bf16
  u16*   qbv = (u16*)alloc(M * FF * 2);        // q, bf16
  u16*   kbv = (u16*)alloc(M * FF * 2);        // k, bf16
  float* ao  = (float*)alloc(M * FF * 4);      // attn_out, f32
  u16*   obf = (u16*)alloc(M * FF * 2);        // gated o, bf16
  (void)ws_size; (void)in_sizes; (void)n_in; (void)out_size;

  k_ln_x<<<dim3((unsigned)M), dim3(256), 0, stream>>>(x, tl, xn);
  k_tr_w<<<dim3((NOUT1 * KTOT + 255) / 256), dim3(256), 0, stream>>>(w1, wt1, KTOT, NOUT1);
  k_tr_w<<<dim3((FF * KTOT + 255) / 256), dim3(256), 0, stream>>>(w2, wt2, KTOT, FF);
  k_gemm_uvqk<<<dim3(128, 32), dim3(256), 0, stream>>>(xn, wt1, tl, uu, vt, qbv, kbv);
  k_attn<<<dim3(4, HH, BB), dim3(256), 0, stream>>>(qbv, kbv, vt, tl, tw, pw, ts, ao);
  k_ln_attn<<<dim3((unsigned)M), dim3(256), 0, stream>>>(ao, uu, tl, obf);
  k_gemm_out<<<dim3(128, 8), dim3(256), 0, stream>>>(obf, wt2, x, obv, (float*)d_out);
}
